// MultiHeadAttention_70884140253264
// MI455X (gfx1250) — matrix operand order repaired and checked by reading, not compiled, not run on hardware
//
#include <hip/hip_runtime.h>
#include <math.h>

#define DIM    1024
#define NHEADS 8
#define BB     2
#define SS     2048

typedef __attribute__((ext_vector_type(16))) __bf16       v16bf;
typedef __attribute__((ext_vector_type(8)))  float        v8f;
typedef __attribute__((ext_vector_type(4))) unsigned int  u32x4;
typedef __attribute__((ext_vector_type(8)))  int          i32x8;
typedef __attribute__((ext_vector_type(4)))  int          i32x4;
typedef __attribute__((ext_vector_type(4)))  float        f32x4;

union FragU { v16bf v; unsigned int u[8]; };

#define LDS_ROW 40   // bf16 per LDS tile row: 32 payload + 8 pad (80B stride)

#if __has_builtin(__builtin_amdgcn_tensor_load_to_lds)
#define HAVE_TDM 1
#else
#define HAVE_TDM 0
#endif

// raw LDS byte offset of a __shared__ pointer (generic -> AS(3) -> int)
__device__ __forceinline__ unsigned lds_addr_of(const void* p) {
  typedef __attribute__((address_space(3))) const void* lds_cptr;
  return (unsigned)(unsigned long long)(lds_cptr)p;
}

// TDM: DMA a 128-row x 32-col bf16 tile (row stride K elems) into LDS with
// 16-dword-interval / 4-dword padding => 80-byte LDS row stride.
__device__ __forceinline__ void tdm_load_tile(unsigned lds_byte, const __bf16* gsrc, int K) {
#if HAVE_TDM
  unsigned long long ga = (unsigned long long)gsrc;
  u32x4 g0;
  g0[0] = 1u;                                   // count=1 (valid descriptor)
  g0[1] = lds_byte;                             // lds_addr
  g0[2] = (unsigned)ga;                         // global_addr[31:0]
  g0[3] = (unsigned)(ga >> 32) | (2u << 30);    // global_addr[56:32] | type=2
  i32x8 g1;
  g1[0] = (1 << 16) | (1 << 20) | (3 << 22) | (3 << 25); // data_size=2B, pad_en, interval=16dw, amount=4dw
  g1[1] = (K & 0xffff) << 16;                   // tensor_dim0 = K (low 16)
  g1[2] = (int)(0xffffu << 16);                 // tensor_dim0 hi=0, tensor_dim1 lo16 = 0xffff (no clip)
  g1[3] = 32 << 16;                             // tensor_dim1 hi=0, tile_dim0 = 32
  g1[4] = 128;                                  // tile_dim1 = 128, tile_dim2 = 0
  g1[5] = K;                                    // tensor_dim0_stride (data_size units)
  g1[6] = 0;
  g1[7] = 0;
  i32x4 z4 = {0, 0, 0, 0};
  i32x8 z8 = {0, 0, 0, 0, 0, 0, 0, 0};
  __builtin_amdgcn_tensor_load_to_lds(g0, g1, z4, z4, z8, 0);
#endif
}

// ---------------- f32 -> bf16 elementwise ----------------
__global__ void k_f32_to_bf16(const float* __restrict__ in, __bf16* __restrict__ out, int n) {
  int i = blockIdx.x * blockDim.x + threadIdx.x;
  int stride = gridDim.x * blockDim.x;
  for (; i < n; i += stride) out[i] = (__bf16)in[i];
}

// Wo_eff[o][d] = sum_h Wo[o][h*DIM + d]
__global__ void k_reduce_wo(const float* __restrict__ Wo, __bf16* __restrict__ out) {
  int i = blockIdx.x * blockDim.x + threadIdx.x;
  if (i >= DIM * DIM) return;
  int o = i >> 10, d = i & (DIM - 1);
  float s = 0.f;
#pragma unroll
  for (int h = 0; h < NHEADS; ++h) s += Wo[(size_t)o * (DIM * NHEADS) + h * DIM + d];
  out[i] = (__bf16)s;
}

// transpose [B,S,D] -> [B,D,S] (bf16)
__global__ void k_transpose_bf16(const __bf16* __restrict__ in, __bf16* __restrict__ out) {
  __shared__ __bf16 tile[32][33];
  int tx = threadIdx.x & 31, ty = threadIdx.x >> 5;
  int d0 = blockIdx.x * 32, s0 = blockIdx.y * 32, b = blockIdx.z;
#pragma unroll
  for (int r = 0; r < 4; ++r)
    tile[ty + 8 * r][tx] = in[((size_t)b * SS + s0 + ty + 8 * r) * DIM + d0 + tx];
  __syncthreads();
#pragma unroll
  for (int r = 0; r < 4; ++r)
    out[((size_t)b * DIM + d0 + ty + 8 * r) * SS + s0 + tx] = tile[tx][ty + 8 * r];
}

// row softmax over head-0 scores, + 1e-5, replicated into all 8 head slots
__global__ __launch_bounds__(256) void k_softmax(float* __restrict__ att) {
  const int s = blockIdx.x, b = blockIdx.y, t = threadIdx.x;
  const float* row0 = att + (((size_t)b * NHEADS) * SS + s) * SS;
  float x[8];
  float mx = -3.4e38f;
#pragma unroll
  for (int i = 0; i < 8; ++i) { x[i] = row0[t + i * 256]; mx = fmaxf(mx, x[i]); }
  __shared__ float red[256];
  red[t] = mx; __syncthreads();
  for (int off = 128; off > 0; off >>= 1) {
    if (t < off) red[t] = fmaxf(red[t], red[t + off]);
    __syncthreads();
  }
  mx = red[0]; __syncthreads();
  float sum = 0.f;
#pragma unroll
  for (int i = 0; i < 8; ++i) { x[i] = __expf(x[i] - mx); sum += x[i]; }
  red[t] = sum; __syncthreads();
  for (int off = 128; off > 0; off >>= 1) {
    if (t < off) red[t] += red[t + off];
    __syncthreads();
  }
  const float inv = 1.f / red[0];
#pragma unroll
  for (int i = 0; i < 8; ++i) {
    const float v = x[i] * inv + 1e-5f;
#pragma unroll
    for (int h = 0; h < NHEADS; ++h)
      att[(((size_t)b * NHEADS + h) * SS + s) * SS + t + i * 256] = v;
  }
}

// ---------------- bf16 WMMA GEMM: C[M,N] = scale*(A[M,K] @ Bmat[N,K]^T) (+bias) ----------------
// 128x128x32 block tile, double-buffered LDS, TDM (wave 0) feeds bf16 tiles,
// f32 A staged manually (register-pipelined) with on-the-fly bf16 conversion.
template <bool A_IS_F32>
__global__ __launch_bounds__(256)
void k_gemm_wmma(const void* __restrict__ Aptr, const __bf16* __restrict__ Bmat,
                 float* __restrict__ outF, __bf16* __restrict__ outB,
                 const float* __restrict__ bias,
                 int N, int K, long sA, long sB, long sOF, long sOB, float scale) {
  __shared__ __bf16 As[2][128 * LDS_ROW];
  __shared__ __bf16 Bs[2][128 * LDS_ROW];

  const int t    = threadIdx.x;
  const int lane = t & 31;
  const int wave = t >> 5;
  const int wr   = wave >> 2;   // 0..1 : 64-row slab
  const int wc   = wave & 3;    // 0..3 : 32-col slab
  const int l16  = lane & 15;
  const int lh   = lane >> 4;
  const int r0   = t >> 2;      // 0..63 (stages rows r0, r0+64)
  const int c8   = (t & 3) * 8;
  const int mBase = blockIdx.y * 128;
  const int nBase = blockIdx.x * 128;

  const size_t aOff  = (size_t)blockIdx.z * (size_t)sA;
  const __bf16* Bb   = Bmat + (size_t)blockIdx.z * (size_t)sB;
  const __bf16* Ab16 = (const __bf16*)Aptr + aOff;
  const float*  Af32 = (const float*)Aptr + aOff;

  constexpr bool TDM_B = HAVE_TDM != 0;
  constexpr bool TDM_A = (HAVE_TDM != 0) && !A_IS_F32;

  v8f acc[4][2] = {};

  // ---- prologue: stage tile 0 into buffer 0 ----
  if (wave == 0) {
    if (TDM_B) tdm_load_tile(lds_addr_of(&Bs[0][0]), Bb + (size_t)nBase * K, K);
    if (TDM_A) tdm_load_tile(lds_addr_of(&As[0][0]), Ab16 + (size_t)mBase * K, K);
  }
  if (!TDM_B) {
#pragma unroll
    for (int i = 0; i < 2; ++i)
      *(u32x4*)&Bs[0][(r0 + 64 * i) * LDS_ROW + c8] =
          *(const u32x4*)(Bb + (size_t)(nBase + r0 + 64 * i) * K + c8);
  }
  if (!TDM_A) {
#pragma unroll
    for (int i = 0; i < 2; ++i) {
      if (A_IS_F32) {
        const float* p = Af32 + (size_t)(mBase + r0 + 64 * i) * K + c8;
        f32x4 a0 = ((const f32x4*)p)[0];
        f32x4 a1 = ((const f32x4*)p)[1];
        union { __bf16 h[8]; u32x4 v; } cv;
#pragma unroll
        for (int j = 0; j < 4; ++j) { cv.h[j] = (__bf16)a0[j]; cv.h[4 + j] = (__bf16)a1[j]; }
        *(u32x4*)&As[0][(r0 + 64 * i) * LDS_ROW + c8] = cv.v;
      } else {
        *(u32x4*)&As[0][(r0 + 64 * i) * LDS_ROW + c8] =
            *(const u32x4*)(Ab16 + (size_t)(mBase + r0 + 64 * i) * K + c8);
      }
    }
  }
#if HAVE_TDM
  if (wave == 0) __builtin_amdgcn_s_wait_tensorcnt(0);
#endif
  __syncthreads();

  const int nk = K / 32;
  for (int kt = 0; kt < nk; ++kt) {
    const int cur  = kt & 1;
    const int nxt  = cur ^ 1;
    const int k0n  = (kt + 1) * 32;
    const bool more = (kt + 1 < nk);

    // ---- issue next-tile TDM DMAs early (overlap with WMMA below) ----
    if (more && wave == 0) {
      if (TDM_B) tdm_load_tile(lds_addr_of(&Bs[nxt][0]), Bb + (size_t)nBase * K + k0n, K);
      if (TDM_A) tdm_load_tile(lds_addr_of(&As[nxt][0]), Ab16 + (size_t)mBase * K + k0n, K);
    }
    // ---- issue next-tile global loads into registers (manual paths) ----
    f32x4 pa0[2], pa1[2];
    u32x4 pa16[2], pb16[2];
    if (more) {
      if (A_IS_F32) {
#pragma unroll
        for (int i = 0; i < 2; ++i) {
          const float* p = Af32 + (size_t)(mBase + r0 + 64 * i) * K + k0n + c8;
          pa0[i] = ((const f32x4*)p)[0];
          pa1[i] = ((const f32x4*)p)[1];
        }
      } else if (!TDM_A) {
#pragma unroll
        for (int i = 0; i < 2; ++i)
          pa16[i] = *(const u32x4*)(Ab16 + (size_t)(mBase + r0 + 64 * i) * K + k0n + c8);
      }
      if (!TDM_B) {
#pragma unroll
        for (int i = 0; i < 2; ++i)
          pb16[i] = *(const u32x4*)(Bb + (size_t)(nBase + r0 + 64 * i) * K + k0n + c8);
      }
    }

    // ---- fragments from current buffer (ISA VGPR layouts) ----
    FragU af[4];
#pragma unroll
    for (int mt = 0; mt < 4; ++mt) {
      const int arow = wr * 64 + mt * 16 + l16;
      const int kb   = lh * 8;
#pragma unroll
      for (int v = 0; v < 8; ++v) {
        const int kk = kb + (v & 3) * 2 + (v >> 2) * 16;
        af[mt].u[v] = *(const unsigned int*)&As[cur][arow * LDS_ROW + kk];
      }
    }
    FragU bfm[2];
#pragma unroll
    for (int nt = 0; nt < 2; ++nt) {
      const int bn = wc * 32 + nt * 16 + l16;
      const int kb = lh * 8;
#pragma unroll
      for (int v = 0; v < 8; ++v)
        bfm[nt].u[v] = *(const unsigned int*)&Bs[cur][bn * LDS_ROW + kb + (v & 3) * 2 + (v >> 2) * 16];
    }

#pragma unroll
    for (int mt = 0; mt < 4; ++mt)
#pragma unroll
      for (int nt = 0; nt < 2; ++nt)
        acc[mt][nt] = __builtin_amdgcn_wmma_f32_16x16x32_bf16(
            false, af[mt].v, false, bfm[nt].v, (short)0, acc[mt][nt], false, false);

    // ---- commit staged registers into the next buffer ----
    if (more) {
      if (A_IS_F32) {
#pragma unroll
        for (int i = 0; i < 2; ++i) {
          union { __bf16 h[8]; u32x4 v; } cv;
#pragma unroll
          for (int j = 0; j < 4; ++j) { cv.h[j] = (__bf16)pa0[i][j]; cv.h[4 + j] = (__bf16)pa1[i][j]; }
          *(u32x4*)&As[nxt][(r0 + 64 * i) * LDS_ROW + c8] = cv.v;
        }
      } else if (!TDM_A) {
#pragma unroll
        for (int i = 0; i < 2; ++i)
          *(u32x4*)&As[nxt][(r0 + 64 * i) * LDS_ROW + c8] = pa16[i];
      }
      if (!TDM_B) {
#pragma unroll
        for (int i = 0; i < 2; ++i)
          *(u32x4*)&Bs[nxt][(r0 + 64 * i) * LDS_ROW + c8] = pb16[i];
      }
    }
#if HAVE_TDM
    if (wave == 0) __builtin_amdgcn_s_wait_tensorcnt(0);
#endif
    __syncthreads();
  }

  // ---- epilogue ----
#pragma unroll
  for (int mt = 0; mt < 4; ++mt) {
#pragma unroll
    for (int nt = 0; nt < 2; ++nt) {
      const int col  = nBase + wc * 32 + nt * 16 + l16;
      const float bv = bias ? bias[col] : 0.f;
#pragma unroll
      for (int r = 0; r < 8; ++r) {
        const int row = mBase + wr * 64 + mt * 16 + lh * 8 + r;
        const float val = acc[mt][nt][r] * scale + bv;
        if (outF) outF[(size_t)blockIdx.z * sOF + (size_t)row * N + col] = val;
        if (outB) outB[(size_t)blockIdx.z * sOB + (size_t)row * N + col] = (__bf16)val;
      }
    }
  }
}

extern "C" void kernel_launch(void* const* d_in, const int* in_sizes, int n_in,
                              void* d_out, int out_size, void* d_ws, size_t ws_size,
                              hipStream_t stream) {
  const float* q  = (const float*)d_in[0];   // k (d_in[1]) / v (d_in[2]) unused by the reference
  const float* Wq = (const float*)d_in[3];
  const float* Wk = (const float*)d_in[4];
  const float* Wv = (const float*)d_in[5];
  const float* Wo = (const float*)d_in[6];
  const float* bo = (const float*)d_in[7];

  float* x_out   = (float*)d_out;                   // [B,S,DIM]
  float* att_out = x_out + (size_t)BB * SS * DIM;   // [B,8,S,S]

  char* w = (char*)d_ws;
  const size_t MB = (size_t)1 << 20;
  __bf16* Wqb  = (__bf16*)(w + 0 * MB);
  __bf16* Wkb  = (__bf16*)(w + 2 * MB);
  __bf16* Wvb  = (__bf16*)(w + 4 * MB);
  __bf16* WoEb = (__bf16*)(w + 6 * MB);
  __bf16* qp   = (__bf16*)(w + 8 * MB);   // [B*S, DIM]
  __bf16* kp   = (__bf16*)(w + 16 * MB);  // [B*S, DIM]
  __bf16* vp   = (__bf16*)(w + 24 * MB);  // [B*S, DIM]
  __bf16* vpT  = (__bf16*)(w + 32 * MB);  // [B, DIM, S]
  __bf16* xh   = (__bf16*)(w + 40 * MB);  // [B*S, DIM]

  k_f32_to_bf16<<<1024, 256, 0, stream>>>(Wq, Wqb, DIM * DIM);
  k_f32_to_bf16<<<1024, 256, 0, stream>>>(Wk, Wkb, DIM * DIM);
  k_f32_to_bf16<<<1024, 256, 0, stream>>>(Wv, Wvb, DIM * DIM);
  k_reduce_wo<<<(DIM * DIM) / 256, 256, 0, stream>>>(Wo, WoEb);

  dim3 gp(DIM / 128, (BB * SS) / 128, 1);
  k_gemm_wmma<true><<<gp, 256, 0, stream>>>(q, Wqb, (float*)nullptr, qp, (const float*)nullptr,
                                            DIM, DIM, 0L, 0L, 0L, 0L, 1.f);
  k_gemm_wmma<true><<<gp, 256, 0, stream>>>(q, Wkb, (float*)nullptr, kp, (const float*)nullptr,
                                            DIM, DIM, 0L, 0L, 0L, 0L, 1.f);
  k_gemm_wmma<true><<<gp, 256, 0, stream>>>(q, Wvb, (float*)nullptr, vp, (const float*)nullptr,
                                            DIM, DIM, 0L, 0L, 0L, 0L, 1.f);

  k_transpose_bf16<<<dim3(DIM / 32, SS / 32, BB), 256, 0, stream>>>(vp, vpT);

  dim3 gs(SS / 128, SS / 128, BB);
  k_gemm_wmma<false><<<gs, 256, 0, stream>>>(qp, kp, att_out, (__bf16*)nullptr, (const float*)nullptr,
                                             SS, DIM, (long)SS * DIM, (long)SS * DIM,
                                             (long)NHEADS * SS * SS, 0L, 1.f / 320.f);

  k_softmax<<<dim3(SS, BB), 256, 0, stream>>>(att_out);

  dim3 gx(DIM / 128, SS / 128, BB);
  k_gemm_wmma<true><<<gx, 256, 0, stream>>>(att_out, vpT, (float*)nullptr, xh, (const float*)nullptr,
                                            DIM, SS, (long)NHEADS * SS * SS, (long)DIM * SS,
                                            0L, (long)SS * DIM, 1.f);

  dim3 gf(DIM / 128, (BB * SS) / 128, 1);
  k_gemm_wmma<false><<<gf, 256, 0, stream>>>(xh, WoEb, x_out, (__bf16*)nullptr, bo,
                                             DIM, DIM, 0L, 0L, 0L, 0L, 1.f);
}
